// SpatialAnomalyScorer_60352880443735
// MI455X (gfx1250) — matrix operand order repaired and checked by reading, not compiled, not run on hardware
//
#include <hip/hip_runtime.h>
#include <hip/hip_bf16.h>

typedef __attribute__((ext_vector_type(16))) _Float16 v16h;
typedef __attribute__((ext_vector_type(8)))  float    v8f;
typedef int v4i __attribute__((vector_size(16)));

#define DEV static __device__ __forceinline__

// ---------------------------------------------------------------------------
// gfx1250 async global->LDS copy (ASYNCcnt path), guarded so the file always
// compiles: falls back to a synchronous VGPR round-trip if the builtins are
// absent from this toolchain. Builtin prototype (from clang diagnostics):
//   (as1 v4i* global_src, as3 v4i* lds_dst, imm offset, imm cpol)
// ---------------------------------------------------------------------------
#if defined(__has_builtin)
#if __has_builtin(__builtin_amdgcn_global_load_async_to_lds_b128) && \
    __has_builtin(__builtin_amdgcn_s_wait_asynccnt)
#define USE_ASYNC_LDS 1
#endif
#endif
#ifndef USE_ASYNC_LDS
#define USE_ASYNC_LDS 0
#endif

DEV void cp16_g2l(const uint4* g, uint4* l) {
#if USE_ASYNC_LDS
  __builtin_amdgcn_global_load_async_to_lds_b128(
      (__attribute__((address_space(1))) v4i*)g,
      (__attribute__((address_space(3))) v4i*)l, 0, 0);
#else
  *l = *g;
#endif
}

template <int N>
DEV void async_wait() {
#if USE_ASYNC_LDS
  __builtin_amdgcn_s_wait_asynccnt(N);
#endif
}

DEV unsigned int pack2(float x, float y) {
  _Float16 lo = (_Float16)x, hi = (_Float16)y;
  unsigned short a, b;
  __builtin_memcpy(&a, &lo, 2);
  __builtin_memcpy(&b, &hi, 2);
  return (unsigned int)a | ((unsigned int)b << 16);
}

DEV float sigmoidf(float x) { return 1.0f / (1.0f + __expf(-x)); }

// ---------------------------------------------------------------------------
// B-fragment-friendly packing: index (kc, hi, col, v) -> 8 contiguous dwords
// per lane, so each WMMA B operand is two global_load_b128.
//   kk (K-pair index) = kc*16 + hi*8 + v
// ---------------------------------------------------------------------------

// prep_protos: f32 [1000][512] -> pp[((kc*2+hi)*1008 + j)*8 + v] + pnorm[1008]
__global__ __launch_bounds__(256) void k_prep_protos(
    const float* __restrict__ proto, unsigned int* __restrict__ pp,
    float* __restrict__ pnorm) {
  __shared__ float red[256];
  int j = blockIdx.x;   // 0..1007
  int t = threadIdx.x;  // kk = t
  float a = 0.f, b = 0.f;
  if (j < 1000) {
    a = proto[(size_t)j * 512 + 2 * t];
    b = proto[(size_t)j * 512 + 2 * t + 1];
  }
  int kc = t >> 4, hi = (t >> 3) & 1, v = t & 7;
  pp[((size_t)(kc * 2 + hi) * 1008 + j) * 8 + v] = pack2(a, b);
  red[t] = a * a + b * b;
  __syncthreads();
  for (int s = 128; s > 0; s >>= 1) {
    if (t < s) red[t] += red[t + s];
    __syncthreads();
  }
  if (t == 0) pnorm[j] = (j < 1000) ? red[0] : 3.0e38f;
}

// ---------------------------------------------------------------------------
// prep_x: f4 NCHW f32 -> xh pixel-major f16 [32768][512] + xnorm[32768]
// ---------------------------------------------------------------------------
__global__ __launch_bounds__(256) void k_prep_x(
    const float* __restrict__ f4, _Float16* __restrict__ xh,
    float* __restrict__ xnorm) {
  __shared__ _Float16 shx[32 * 512];
  __shared__ float shr[256];
  int t = threadIdx.x;
  int pix0 = blockIdx.x * 32;
  int b = pix0 >> 10;
  int hw0 = pix0 & 1023;
  int p = t & 31, cc = t >> 5;
  float acc = 0.f;
  const float* src = f4 + (size_t)b * 512 * 1024 + hw0;
  for (int i = 0; i < 64; ++i) {
    int c = i * 8 + cc;
    float v = src[(size_t)c * 1024 + p];
    shx[p * 512 + c] = (_Float16)v;
    acc += v * v;
  }
  shr[t] = acc;
  __syncthreads();
  if (t < 32) {
    float s = 0.f;
    for (int k = 0; k < 8; ++k) s += shr[k * 32 + t];
    xnorm[pix0 + t] = s;
  }
  const uint4* s4 = (const uint4*)shx;
  uint4* d4 = (uint4*)(xh + (size_t)pix0 * 512);
  for (int i = t; i < 2048; i += 256) d4[i] = s4[i];
}

// ---------------------------------------------------------------------------
// pack_w: w [O][CIN][3][3] -> wp[(((tap*(CIN/32)+kc)*2+hi)*O + o)*8 + v]
// ---------------------------------------------------------------------------
template <int O, int CIN>
__global__ __launch_bounds__(256) void k_pack_w(
    const float* __restrict__ w, unsigned int* __restrict__ wp) {
  constexpr int KK = CIN / 2;
  int idx = blockIdx.x * 256 + threadIdx.x;
  if (idx >= 9 * KK * O) return;
  int o = idx % O;
  int kk = (idx / O) % KK;
  int tap = idx / (O * KK);
  float a = w[((size_t)o * CIN + 2 * kk) * 9 + tap];
  float b = w[((size_t)o * CIN + 2 * kk + 1) * 9 + tap];
  int kc = kk >> 4, hi = (kk >> 3) & 1, v = kk & 7;
  wp[(((size_t)(tap * (CIN / 32) + kc) * 2 + hi) * O + o) * 8 + v] =
      pack2(a, b);
}

// ---------------------------------------------------------------------------
// dist_min: 16 pixels x 1008 prototypes per block, WMMA f16->f32.
// Per-lane running min across tiles; single cross-lane reduce at the end.
// ---------------------------------------------------------------------------
__global__ __launch_bounds__(256) void k_dist(
    const _Float16* __restrict__ xh, const float* __restrict__ xnorm,
    const unsigned int* __restrict__ pp, const float* __restrict__ pnorm,
    float* __restrict__ sdist) {
  __shared__ _Float16 xt[16 * 512];
  __shared__ float xn[16];
  __shared__ int dmin[16];
  int t = threadIdx.x;
  int row0 = blockIdx.x * 16;
  {  // async-stage the 16KB pixel tile into LDS
    const uint4* s4 = (const uint4*)(xh + (size_t)row0 * 512);
    uint4* d4 = (uint4*)xt;
#pragma unroll
    for (int i = 0; i < 4; ++i) cp16_g2l(s4 + t + i * 256, d4 + t + i * 256);
  }
  if (t < 16) {
    xn[t] = xnorm[row0 + t];
    dmin[t] = 0x7f7fffff;  // +FLT_MAX bits
  }
  async_wait<0>();
  __syncthreads();
  int wave = t >> 5, lane = t & 31;
  int m = lane & 15, hi = lane >> 4, n = m;
  float xnv[8];
#pragma unroll
  for (int v = 0; v < 8; ++v) xnv[v] = xn[v + 8 * hi];
  float rm[8];
#pragma unroll
  for (int v = 0; v < 8; ++v) rm[v] = 3.4e38f;

  for (int jt = wave; jt < 63; jt += 8) {
    v8f acc = {};
    int proto = jt * 16 + n;
    for (int kc = 0; kc < 16; ++kc) {
      v16h a, bf;
      uint4* au4 = (uint4*)&a;
      uint4* bu4 = (uint4*)&bf;
      const uint4* ap = (const uint4*)(xt + m * 512 + kc * 32 + 16 * hi);
      au4[0] = ap[0];
      au4[1] = ap[1];  // +8 halfs
      const uint4* bp =
          (const uint4*)(pp + ((size_t)(kc * 2 + hi) * 1008 + proto) * 8);
      bu4[0] = bp[0];
      bu4[1] = bp[1];
      acc = __builtin_amdgcn_wmma_f32_16x16x32_f16(false, a, false, bf,
                                                   (short)0, acc, false, false);
    }
    float pn = pnorm[proto];
#pragma unroll
    for (int v = 0; v < 8; ++v)
      rm[v] = fminf(rm[v], fmaxf(xnv[v] + pn - 2.0f * acc[v], 0.0f));
  }
  // one cross-lane min per accumulator row, then LDS atomic across waves
#pragma unroll
  for (int v = 0; v < 8; ++v) {
    float d = rm[v];
#pragma unroll
    for (int msk = 8; msk >= 1; msk >>= 1)
      d = fminf(d, __shfl_xor(d, msk, 32));
    if (m == 0) atomicMin(&dmin[v + 8 * hi], __float_as_int(d));
  }
  __syncthreads();
  if (t < 16) {
    float md = sqrtf(__int_as_float(dmin[t]));
    sdist[row0 + t] = sigmoidf(md - 2.0f);
  }
}

// ---------------------------------------------------------------------------
// conv_igemm: implicit-GEMM 3x3 conv (pad 1) via WMMA; 16 pixels x COUT/block.
// Double-buffered LDS A-tile: tap t+1 is staged (async when available) while
// tap t's WMMAs run. Sources are edge-clamped so every wave issues a uniform
// number of async ops; halo rows are zeroed afterwards with DS stores.
// ---------------------------------------------------------------------------
template <int CIN, int COUT>
__global__ __launch_bounds__(256) void k_conv(
    const _Float16* __restrict__ xin, const unsigned int* __restrict__ wp,
    const float* __restrict__ bias, float* __restrict__ y) {
  constexpr int NT = COUT / 16;
  constexpr int PW = (NT + 7) / 8;
  constexpr int VEC = CIN / 128;  // uint4 per (row, sub-thread)
  __shared__ _Float16 at[2][16 * CIN];
  int t = threadIdx.x;
  int pix0 = blockIdx.x * 16;
  int b = pix0 >> 10;
  int y0 = (pix0 & 1023) >> 5;
  int x0 = pix0 & 31;
  int wave = t >> 5, lane = t & 31;
  int m = lane & 15, hi = lane >> 4, n = m;
  int lrow = t >> 4, sub = t & 15;

  v8f zero8 = {};
  v8f acc[PW];
#pragma unroll
  for (int i = 0; i < PW; ++i) acc[i] = zero8;

  // stage tap into LDS buffer buf (clamped source; uniform issue count)
  auto stage = [&](int buf, int tap) {
    int dy = tap / 3 - 1, dx = tap % 3 - 1;
    int yy = y0 + dy, xx = x0 + lrow + dx;
    int yyc = min(max(yy, 0), 31), xxc = min(max(xx, 0), 31);
    const uint4* src =
        (const uint4*)(xin + (size_t)(b * 1024 + yyc * 32 + xxc) * CIN) +
        sub * VEC;
    uint4* dst = (uint4*)(at[buf] + lrow * CIN) + sub * VEC;
#pragma unroll
    for (int i = 0; i < VEC; ++i) cp16_g2l(src + i, dst + i);
  };
  // zero the halo rows of buffer buf for tap (after its copies completed)
  auto fix_halo = [&](int buf, int tap) {
    int dy = tap / 3 - 1, dx = tap % 3 - 1;
    int yy = y0 + dy, xx = x0 + lrow + dx;
    if (!(yy >= 0 && yy < 32 && xx >= 0 && xx < 32)) {
      uint4 z = {0u, 0u, 0u, 0u};
      uint4* dst = (uint4*)(at[buf] + lrow * CIN) + sub * VEC;
#pragma unroll
      for (int i = 0; i < VEC; ++i) dst[i] = z;
    }
  };

  stage(0, 0);
  for (int tap = 0; tap < 9; ++tap) {
    int cur = tap & 1;
    if (tap < 8) {
      stage(cur ^ 1, tap + 1);  // overlap next tile with this tap's WMMAs
      async_wait<VEC>();        // current tile complete; next may be in flight
    } else {
      async_wait<0>();
    }
    fix_halo(cur, tap);
    __syncthreads();
#pragma unroll
    for (int jl = 0; jl < PW; ++jl) {
      int jt = wave + jl * 8;
      if (jt >= NT) break;
      int o = jt * 16 + n;
      for (int kc = 0; kc < CIN / 32; ++kc) {
        v16h a, bf;
        uint4* au4 = (uint4*)&a;
        uint4* bu4 = (uint4*)&bf;
        const uint4* ap =
            (const uint4*)(at[cur] + m * CIN + kc * 32 + 16 * hi);
        au4[0] = ap[0];
        au4[1] = ap[1];
        const uint4* bp =
            (const uint4*)(wp +
                           (((size_t)(tap * (CIN / 32) + kc) * 2 + hi) * COUT +
                            o) * 8);
        bu4[0] = bp[0];
        bu4[1] = bp[1];
        acc[jl] = __builtin_amdgcn_wmma_f32_16x16x32_f16(
            false, a, false, bf, (short)0, acc[jl], false, false);
      }
    }
    __syncthreads();  // protect buffer reuse on the next iteration
  }
#pragma unroll
  for (int jl = 0; jl < PW; ++jl) {
    int jt = wave + jl * 8;
    if (jt >= NT) break;
    int o = jt * 16 + n;
    float bv = bias[o];
#pragma unroll
    for (int v = 0; v < 8; ++v) {
      int pm = v + 8 * hi;
      y[(size_t)(pix0 + pm) * COUT + o] = acc[jl][v] + bv;
    }
  }
}

// ---------------------------------------------------------------------------
// GroupNorm stats: per (b,g) mean & inv-std over CG channels x 1024 pixels
// ---------------------------------------------------------------------------
template <int COUT, int G>
__global__ __launch_bounds__(256) void k_gnstats(
    const float* __restrict__ y, float* __restrict__ stats) {
  constexpr int CG = COUT / G;
  __shared__ float rs[256], rq[256];
  int bg = blockIdx.x;
  int b = bg / G, g = bg % G;
  int t = threadIdx.x;
  float s = 0.f, q = 0.f;
  for (int idx = t; idx < 1024 * CG; idx += 256) {
    int c = idx % CG, p = idx / CG;
    float v = y[(size_t)(b * 1024 + p) * COUT + g * CG + c];
    s += v;
    q += v * v;
  }
  rs[t] = s;
  rq[t] = q;
  __syncthreads();
  for (int st = 128; st > 0; st >>= 1) {
    if (t < st) {
      rs[t] += rs[t + st];
      rq[t] += rq[t + st];
    }
    __syncthreads();
  }
  if (t == 0) {
    float inv_n = 1.0f / (1024.f * CG);
    float mean = rs[0] * inv_n;
    float var = rq[0] * inv_n - mean * mean;
    stats[bg * 2 + 0] = mean;
    stats[bg * 2 + 1] = rsqrtf(var + 1e-5f);
  }
}

// ---------------------------------------------------------------------------
// GroupNorm apply + ReLU + f16 cast (pixel-major out)
// ---------------------------------------------------------------------------
template <int COUT, int G>
__global__ __launch_bounds__(256) void k_gnapply(
    const float* __restrict__ y, const float* __restrict__ stats,
    const float* __restrict__ gamma, const float* __restrict__ beta,
    _Float16* __restrict__ h) {
  constexpr int CG = COUT / G;
  size_t idx = (size_t)blockIdx.x * 256 + threadIdx.x;
  int c = (int)(idx % COUT);
  int pix = (int)(idx / COUT);
  int b = pix >> 10;
  int g = c / CG;
  float mean = stats[(b * G + g) * 2 + 0];
  float inv = stats[(b * G + g) * 2 + 1];
  float v = (y[idx] - mean) * inv * gamma[c] + beta[c];
  h[idx] = (_Float16)fmaxf(v, 0.f);
}

// ---------------------------------------------------------------------------
// conv3 (1x1, 128->1) + sigmoid + fuse with distance score. Wave per pixel.
// ---------------------------------------------------------------------------
__global__ __launch_bounds__(256) void k_conv3(
    const _Float16* __restrict__ h2, const float* __restrict__ w3,
    const float* __restrict__ b3, const float* __restrict__ sdist,
    float* __restrict__ sfused) {
  int t = threadIdx.x;
  int lane = t & 31;
  int pix = blockIdx.x * 8 + (t >> 5);
  float acc = 0.f;
#pragma unroll
  for (int i = 0; i < 4; ++i) {
    int c = lane + 32 * i;
    acc += (float)h2[(size_t)pix * 128 + c] * w3[c];
  }
#pragma unroll
  for (int msk = 16; msk >= 1; msk >>= 1) acc += __shfl_xor(acc, msk, 32);
  if (lane == 0) {
    float sl = sigmoidf(acc + b3[0]);
    sfused[pix] = 0.5f * (sdist[pix] + sl);
  }
}

// ---------------------------------------------------------------------------
// multiscale: 3x3 + 5x5 + 7x7 single-channel convs, averaged, clipped
// ---------------------------------------------------------------------------
__global__ __launch_bounds__(1024) void k_mscale(
    const float* __restrict__ s, const float* __restrict__ wa1,
    const float* __restrict__ ba1, const float* __restrict__ wa2,
    const float* __restrict__ ba2, const float* __restrict__ wa3,
    const float* __restrict__ ba3, float* __restrict__ out) {
  __shared__ float tile[32 * 32];
  int b = blockIdx.x, t = threadIdx.x;
  tile[t] = s[b * 1024 + t];
  __syncthreads();
  int y = t >> 5, x = t & 31;
  float a1 = ba1[0], a2 = ba2[0], a3 = ba3[0];
  for (int dy = -3; dy <= 3; ++dy) {
    int yy = y + dy;
    if (yy < 0 || yy >= 32) continue;
    for (int dx = -3; dx <= 3; ++dx) {
      int xx = x + dx;
      if (xx < 0 || xx >= 32) continue;
      float v = tile[yy * 32 + xx];
      if (dy >= -1 && dy <= 1 && dx >= -1 && dx <= 1)
        a1 += v * wa1[(dy + 1) * 3 + (dx + 1)];
      if (dy >= -2 && dy <= 2 && dx >= -2 && dx <= 2)
        a2 += v * wa2[(dy + 2) * 5 + (dx + 2)];
      a3 += v * wa3[(dy + 3) * 7 + (dx + 3)];
    }
  }
  float r = (a1 + a2 + a3) * (1.0f / 3.0f);
  out[b * 1024 + t] = fminf(fmaxf(r, 0.f), 1.f);
}

// ---------------------------------------------------------------------------
// bilinear resize 32x32 -> 256x256, half-pixel centers, edge clamp
// ---------------------------------------------------------------------------
__global__ __launch_bounds__(256) void k_resize(
    const float* __restrict__ s, float* __restrict__ out) {
  size_t idx = (size_t)blockIdx.x * 256 + threadIdx.x;
  int x = (int)(idx & 255);
  int y = (int)((idx >> 8) & 255);
  int b = (int)(idx >> 16);
  float fy = (y + 0.5f) * 0.125f - 0.5f;
  float fx = (x + 0.5f) * 0.125f - 0.5f;
  float yf = floorf(fy), xf = floorf(fx);
  int y0 = (int)yf, x0 = (int)xf;
  float wy = fy - yf, wx = fx - xf;
  int y0c = min(max(y0, 0), 31), y1c = min(max(y0 + 1, 0), 31);
  int x0c = min(max(x0, 0), 31), x1c = min(max(x0 + 1, 0), 31);
  const float* sb = s + b * 1024;
  float v00 = sb[y0c * 32 + x0c], v01 = sb[y0c * 32 + x1c];
  float v10 = sb[y1c * 32 + x0c], v11 = sb[y1c * 32 + x1c];
  out[idx] =
      (v00 * (1.f - wx) + v01 * wx) * (1.f - wy) +
      (v10 * (1.f - wx) + v11 * wx) * wy;
}

// ---------------------------------------------------------------------------
// Workspace layout (bytes, 32B-aligned throughout); big buffers aliased.
// ---------------------------------------------------------------------------
constexpr size_t OFF_PP    = 0;                         // 256*1008*4 = 1032192
constexpr size_t OFF_PNORM = 1032192;                   // 1008*4 -> pad 4096
constexpr size_t OFF_XNORM = OFF_PNORM + 4096;          // 32768*4 = 131072
constexpr size_t OFF_W1P   = OFF_XNORM + 131072;        // 9*256*256*4 = 2359296
constexpr size_t OFF_W2P   = OFF_W1P + 2359296;         // 9*128*128*4 = 589824
constexpr size_t OFF_SDIST = OFF_W2P + 589824;          // 131072
constexpr size_t OFF_SFUSE = OFF_SDIST + 131072;        // 131072
constexpr size_t OFF_SAGG  = OFF_SFUSE + 131072;        // 131072
constexpr size_t OFF_GN    = OFF_SAGG + 131072;         // 16384
constexpr size_t OFF_XH    = OFF_GN + 16384;            // 32 MB (y2 aliases)
constexpr size_t OFF_Y1    = OFF_XH + 33554432;         // 32 MB (h2 aliases)
constexpr size_t OFF_H1    = OFF_Y1 + 33554432;         // 16 MB

extern "C" void kernel_launch(void* const* d_in, const int* in_sizes, int n_in,
                              void* d_out, int out_size, void* d_ws,
                              size_t ws_size, hipStream_t stream) {
  (void)in_sizes; (void)n_in; (void)out_size; (void)ws_size;
  const float* f4   = (const float*)d_in[0];
  const float* prot = (const float*)d_in[1];
  const float* w1   = (const float*)d_in[2];
  const float* b1   = (const float*)d_in[3];
  const float* g1   = (const float*)d_in[4];
  const float* be1  = (const float*)d_in[5];
  const float* w2   = (const float*)d_in[6];
  const float* b2   = (const float*)d_in[7];
  const float* g2   = (const float*)d_in[8];
  const float* be2  = (const float*)d_in[9];
  const float* w3   = (const float*)d_in[10];
  const float* b3   = (const float*)d_in[11];
  const float* wa1  = (const float*)d_in[12];
  const float* ba1  = (const float*)d_in[13];
  const float* wa2  = (const float*)d_in[14];
  const float* ba2  = (const float*)d_in[15];
  const float* wa3  = (const float*)d_in[16];
  const float* ba3  = (const float*)d_in[17];

  char* ws = (char*)d_ws;
  unsigned int* pp  = (unsigned int*)(ws + OFF_PP);
  float* pnorm      = (float*)(ws + OFF_PNORM);
  float* xnorm      = (float*)(ws + OFF_XNORM);
  unsigned int* w1p = (unsigned int*)(ws + OFF_W1P);
  unsigned int* w2p = (unsigned int*)(ws + OFF_W2P);
  float* sdist      = (float*)(ws + OFF_SDIST);
  float* sfused     = (float*)(ws + OFF_SFUSE);
  float* sagg       = (float*)(ws + OFF_SAGG);
  float* gns        = (float*)(ws + OFF_GN);
  _Float16* xh      = (_Float16*)(ws + OFF_XH);
  float* y1         = (float*)(ws + OFF_Y1);
  _Float16* h1      = (_Float16*)(ws + OFF_H1);
  float* y2         = (float*)(ws + OFF_XH);    // alias: xh dead after conv1
  _Float16* h2      = (_Float16*)(ws + OFF_Y1); // alias: y1 dead after gn1

  // --- prep ---
  k_prep_protos<<<1008, 256, 0, stream>>>(prot, pp, pnorm);
  k_prep_x<<<1024, 256, 0, stream>>>(f4, xh, xnorm);
  k_pack_w<256, 512><<<2304, 256, 0, stream>>>(w1, w1p);
  k_pack_w<128, 256><<<576, 256, 0, stream>>>(w2, w2p);

  // --- distance branch (GEMM + min) ---
  k_dist<<<2048, 256, 0, stream>>>(xh, xnorm, pp, pnorm, sdist);

  // --- learned branch ---
  k_conv<512, 256><<<2048, 256, 0, stream>>>(xh, w1p, b1, y1);
  k_gnstats<256, 32><<<1024, 256, 0, stream>>>(y1, gns);
  k_gnapply<256, 32><<<32768, 256, 0, stream>>>(y1, gns, g1, be1, h1);
  k_conv<256, 128><<<2048, 256, 0, stream>>>(h1, w2p, b2, y2);
  k_gnstats<128, 16><<<512, 256, 0, stream>>>(y2, gns);
  k_gnapply<128, 16><<<16384, 256, 0, stream>>>(y2, gns, g2, be2, h2);

  // --- fuse + aggregate + resize ---
  k_conv3<<<4096, 256, 0, stream>>>(h2, w3, b3, sdist, sfused);
  k_mscale<<<32, 1024, 0, stream>>>(sfused, wa1, ba1, wa2, ba2, wa3, ba3, sagg);
  k_resize<<<8192, 256, 0, stream>>>(sagg, (float*)d_out);
}
